// SurrogateModel_64450279244337
// MI455X (gfx1250) — compile-verified, matrix operand order checked by reading, not run on hardware
//
#include <hip/hip_runtime.h>
#include <math.h>

#define NN 20000
#define NE 640000
#define F_IN 128
#define F_HID 64
#define F_OUT 32

typedef __attribute__((ext_vector_type(16))) _Float16 v16h;
typedef __attribute__((ext_vector_type(8)))  float    v8f;

// ---- ordered-int mapping for float atomic max ----
__device__ __forceinline__ int fkey(float f) {
  int i = __float_as_int(f);
  return (i >= 0) ? i : (i ^ 0x7FFFFFFF);
}
__device__ __forceinline__ float funkey(int k) {
  return __int_as_float((k >= 0) ? k : (k ^ 0x7FFFFFFF));
}
__device__ __forceinline__ float sigm(float x) { return 1.f / (1.f + expf(-x)); }

__global__ void fill_u32(unsigned int* p, unsigned int v, int n) {
  int i = blockIdx.x * blockDim.x + threadIdx.x;
  if (i < n) p[i] = v;
}

// ------------------------------------------------------------------
// Pre-pack f32 B[K,N] into WMMA B lane layout, f16.
// Packed unit = (kstep, ntile, lane): 16 halves contiguous (32B).
// Lane l (half=l>>4, l16=l&15) of tile (ks,t) holds column t*16+l16,
// K values ks*32 + half*8 + {0..7} and + {16..23}.
// ------------------------------------------------------------------
__global__ void pack_B_f16(const float* __restrict__ B, _Float16* __restrict__ Bp,
                           int Ksteps, int NT, int N) {
  int idx = blockIdx.x * blockDim.x + threadIdx.x;  // Ksteps*NT*32
  if (idx >= Ksteps * NT * 32) return;
  int lane = idx & 31;
  int t    = (idx >> 5) % NT;
  int ks   = (idx >> 5) / NT;
  int half = lane >> 4, l16 = lane & 15;
  int col  = t * 16 + l16;
  int kb   = ks * 32 + half * 8;
  _Float16* o = Bp + (size_t)idx * 16;
#pragma unroll
  for (int j = 0; j < 8; ++j) {
    o[j]     = (_Float16)B[(size_t)(kb + j) * N + col];
    o[j + 8] = (_Float16)B[(size_t)(kb + 16 + j) * N + col];
  }
}

// ------------------------------------------------------------------
// C[M, NT*16] = A[M, K] @ B (pre-packed f16). f32 A converted on the fly,
// f32 accumulate via v_wmma_f32_16x16x32_f16. One wave = 16 rows x all
// NT*16 columns (A fetched once, NT WMMAs per K-step). 2 waves / block.
// ------------------------------------------------------------------
template <int K, int NT>
__global__ void wmma_gemm_f16(const float* __restrict__ A,
                              const _Float16* __restrict__ Bp,
                              float* __restrict__ C) {
  const int lane  = threadIdx.x & 31;
  const int wid   = threadIdx.x >> 5;
  const int mtile = blockIdx.x * 2 + wid;
  const int half  = lane >> 4;
  const int l16   = lane & 15;
  const int rowA  = mtile * 16 + l16;
  constexpr int N = NT * 16;

  v8f acc[NT];
#pragma unroll
  for (int t = 0; t < NT; ++t) acc[t] = (v8f){};

#pragma unroll
  for (int ks = 0; ks < K / 32; ++ks) {
    const float* ap = A + (size_t)rowA * K + ks * 32 + half * 8;
    v16h a;
#pragma unroll
    for (int j = 0; j < 8; ++j) {
      a[j]     = (_Float16)ap[j];
      a[j + 8] = (_Float16)ap[j + 16];
    }
#pragma unroll
    for (int t = 0; t < NT; ++t) {
      const v16h b = *(const v16h*)(Bp + ((size_t)(ks * NT + t) * 32 + lane) * 16);
      acc[t] = __builtin_amdgcn_wmma_f32_16x16x32_f16(false, a, false, b, (short)0,
                                                      acc[t], false, false);
    }
  }
  // C/D layout: VGPR r -> M = mtile*16 + half*8 + r, N = t*16 + l16
  const int mbase = mtile * 16 + half * 8;
#pragma unroll
  for (int t = 0; t < NT; ++t)
#pragma unroll
    for (int r = 0; r < 8; ++r)
      C[(size_t)(mbase + r) * N + t * 16 + l16] = acc[t][r];
}

// es[i] = h[i,:].a_src ; ed[i] = h[i,:].a_dst
__global__ void node_scores(const float* __restrict__ h, const float* __restrict__ a_src,
                            const float* __restrict__ a_dst, float* __restrict__ es,
                            float* __restrict__ ed, int n, int F) {
  int i = blockIdx.x * blockDim.x + threadIdx.x;
  if (i >= n) return;
  float s = 0.f, d = 0.f;
  const float* row = h + (size_t)i * F;
  for (int f = 0; f < F; ++f) { float v = row[f]; s += v * a_src[f]; d += v * a_dst[f]; }
  es[i] = s; ed[i] = d;
}

// pass1: e = leaky_relu(es[src]+ed[dst]); segment max via int-mapped atomicMax
__global__ void edge_pass1(const int* __restrict__ src, const int* __restrict__ dst,
                           const float* __restrict__ es, const float* __restrict__ ed,
                           int* __restrict__ mkey, float* __restrict__ ev, int E, int n) {
  int e = blockIdx.x * blockDim.x + threadIdx.x;
  if (e >= E + n) return;
  int s, d;
  if (e < E) { s = src[e]; d = dst[e]; } else { s = d = e - E; }
  float v = es[s] + ed[d];
  v = (v > 0.f) ? v : 0.2f * v;
  ev[e] = v;
  atomicMax(&mkey[d], fkey(v));
}

// pass2: p = exp(e - m[dst]); denom[dst] += p; store p in place
__global__ void edge_pass2(const int* __restrict__ dst, const int* __restrict__ mkey,
                           float* __restrict__ denom, float* __restrict__ ev, int E, int n) {
  int e = blockIdx.x * blockDim.x + threadIdx.x;
  if (e >= E + n) return;
  int d = (e < E) ? dst[e] : (e - E);
  float p = expf(ev[e] - funkey(mkey[d]));
  ev[e] = p;
  atomicAdd(&denom[d], p);
}

// pass3: agg[dst,f] += p[e] * h[src,f]   (one thread per (edge,feature))
__global__ void edge_agg(const int* __restrict__ src, const int* __restrict__ dst,
                         const float* __restrict__ p, const float* __restrict__ h,
                         float* __restrict__ agg, int E, int n, int fshift) {
  int idx = blockIdx.x * blockDim.x + threadIdx.x;
  int total = (E + n) << fshift;
  if (idx >= total) return;
  int e = idx >> fshift;
  int f = idx & ((1 << fshift) - 1);
  int s, d;
  if (e < E) { s = src[e]; d = dst[e]; } else { s = d = e - E; }
  atomicAdd(&agg[((size_t)d << fshift) + f], p[e] * h[((size_t)s << fshift) + f]);
}

// out = agg/denom + bias (optional relu), in place
__global__ void normalize_bias(float* __restrict__ agg, const float* __restrict__ denom,
                               const float* __restrict__ bias, int n, int fshift,
                               int do_relu) {
  int idx = blockIdx.x * blockDim.x + threadIdx.x;
  if (idx >= (n << fshift)) return;
  int i = idx >> fshift;
  int f = idx & ((1 << fshift) - 1);
  float v = agg[idx] / denom[i] + bias[f];
  if (do_relu) v = fmaxf(v, 0.f);
  agg[idx] = v;
}

// column sums of [n,32] into accum[32]
__global__ void col_sum32(const float* __restrict__ feats, float* __restrict__ accum, int n) {
  int f = threadIdx.x;               // blockDim == 32 (one wave)
  float s = 0.f;
  for (int i = blockIdx.x; i < n; i += gridDim.x) s += feats[(size_t)i * 32 + f];
  atomicAdd(&accum[f], s);
}

// classification head: sigmoid(relu(mean@W1^T+b1)@W2^T+b2) -> out[0]
__global__ void head_c(const float* __restrict__ mean, const float* __restrict__ w1,
                       const float* __restrict__ b1, const float* __restrict__ w2,
                       const float* __restrict__ b2, float* __restrict__ out, float invN) {
  if (threadIdx.x != 0 || blockIdx.x != 0) return;
  float t[32];
  for (int f = 0; f < 32; ++f) t[f] = mean[f] * invN;
  float s2 = b2[0];
  for (int j = 0; j < 16; ++j) {
    float u = b1[j];
    for (int f = 0; f < 32; ++f) u += t[f] * w1[j * 32 + f];
    s2 += fmaxf(u, 0.f) * w2[j];
  }
  out[0] = sigm(s2);
}

// sequential LSTM(32,32) over n steps; out[1] = relu(c_n)@fc1_w^T + fc1_b
__global__ void lstm_seq(const float* __restrict__ feats, const float* __restrict__ W_ih,
                         const float* __restrict__ W_hh, const float* __restrict__ b_ih,
                         const float* __restrict__ b_hh, const float* __restrict__ fc1_w,
                         const float* __restrict__ fc1_b, float* __restrict__ out, int n) {
  __shared__ float h[32], c[32], z[128], xv[32];
  const int r = threadIdx.x;          // 0..127 : gate rows i,f,g,o
  float wih[32], whh[32];
#pragma unroll
  for (int k = 0; k < 32; ++k) { wih[k] = W_ih[r * 32 + k]; whh[k] = W_hh[r * 32 + k]; }
  const float bb = b_ih[r] + b_hh[r];
  if (r < 32) { h[r] = 0.f; c[r] = 0.f; }
  __syncthreads();
  for (int t = 0; t < n; ++t) {
    if (r < 32) xv[r] = feats[(size_t)t * 32 + r];
    __syncthreads();
    float zz = bb;
#pragma unroll
    for (int k = 0; k < 32; ++k) zz += wih[k] * xv[k] + whh[k] * h[k];
    z[r] = zz;
    __syncthreads();
    if (r < 32) {
      float gi = sigm(z[r]);
      float gf = sigm(z[32 + r]);
      float gg = tanhf(z[64 + r]);
      float go = sigm(z[96 + r]);
      float cn = gf * c[r] + gi * gg;
      c[r] = cn;
      h[r] = go * tanhf(cn);
    }
    __syncthreads();
  }
  if (r == 0) {
    float s = fc1_b[0];
    for (int k = 0; k < 32; ++k) s += fmaxf(c[k], 0.f) * fc1_w[k];
    out[1] = s;
  }
}

extern "C" void kernel_launch(void* const* d_in, const int* in_sizes, int n_in,
                              void* d_out, int out_size, void* d_ws, size_t ws_size,
                              hipStream_t stream) {
  const float* x      = (const float*)d_in[0];
  const int*   eidx   = (const int*)d_in[1];
  const int*   esrc   = eidx;
  const int*   edst   = eidx + NE;
  const float* W1     = (const float*)d_in[3];
  const float* a1s    = (const float*)d_in[4];
  const float* a1d    = (const float*)d_in[5];
  const float* b1     = (const float*)d_in[6];
  const float* W2     = (const float*)d_in[7];
  const float* a2s    = (const float*)d_in[8];
  const float* a2d    = (const float*)d_in[9];
  const float* b2     = (const float*)d_in[10];
  const float* W_ih   = (const float*)d_in[11];
  const float* W_hh   = (const float*)d_in[12];
  const float* b_ih   = (const float*)d_in[13];
  const float* b_hh   = (const float*)d_in[14];
  const float* fc1_w  = (const float*)d_in[15];
  const float* fc1_b  = (const float*)d_in[16];
  const float* fc2_w1 = (const float*)d_in[17];
  const float* fc2_b1 = (const float*)d_in[18];
  const float* fc2_w2 = (const float*)d_in[19];
  const float* fc2_b2 = (const float*)d_in[20];
  float* out = (float*)d_out;

  // ---- carve workspace ----
  char* w = (char*)d_ws;
  auto carve = [&](size_t bytes) -> void* {
    void* p = (void*)w;
    w += (bytes + 255) & ~(size_t)255;
    return p;
  };
  float*     h1    = (float*)carve((size_t)NN * F_HID * 4);  // x@W1
  float*     agg1  = (float*)carve((size_t)NN * F_HID * 4);  // layer-1 agg -> feats1
  float*     h2    = (float*)carve((size_t)NN * F_OUT * 4);  // feats1@W2
  float*     agg2  = (float*)carve((size_t)NN * F_OUT * 4);  // layer-2 agg -> feats2
  float*     es    = (float*)carve((size_t)NN * 4);
  float*     ed    = (float*)carve((size_t)NN * 4);
  int*       mkey  = (int*)carve((size_t)NN * 4);
  float*     denom = (float*)carve((size_t)NN * 4);
  float*     ep    = (float*)carve((size_t)(NE + NN) * 4);   // edge e -> p
  float*     meanv = (float*)carve(128);
  _Float16*  Bp1   = (_Float16*)carve((size_t)F_IN * F_HID * 2);  // packed W1
  _Float16*  Bp2   = (_Float16*)carve((size_t)F_HID * F_OUT * 2); // packed W2

  const int EP = NE + NN;                 // edges incl. self loops
  const int TPB = 256;
  auto blks = [](long long n, int t) { return (unsigned)((n + t - 1) / t); };

  // ================= GAT layer 1 =================
  pack_B_f16<<<blks((F_IN / 32) * (F_HID / 16) * 32, TPB), TPB, 0, stream>>>(
      W1, Bp1, F_IN / 32, F_HID / 16, F_HID);
  wmma_gemm_f16<F_IN, F_HID / 16><<<NN / 32, 64, 0, stream>>>(x, Bp1, h1);
  node_scores<<<blks(NN, TPB), TPB, 0, stream>>>(h1, a1s, a1d, es, ed, NN, F_HID);
  fill_u32<<<blks(NN, TPB), TPB, 0, stream>>>((unsigned*)mkey, 0x80000000u, NN);
  fill_u32<<<blks(NN, TPB), TPB, 0, stream>>>((unsigned*)denom, 0u, NN);
  fill_u32<<<blks((long long)NN * F_HID, TPB), TPB, 0, stream>>>((unsigned*)agg1, 0u, NN * F_HID);
  edge_pass1<<<blks(EP, TPB), TPB, 0, stream>>>(esrc, edst, es, ed, mkey, ep, NE, NN);
  edge_pass2<<<blks(EP, TPB), TPB, 0, stream>>>(edst, mkey, denom, ep, NE, NN);
  edge_agg<<<blks((long long)EP * F_HID, TPB), TPB, 0, stream>>>(esrc, edst, ep, h1, agg1, NE, NN, 6);
  normalize_bias<<<blks((long long)NN * F_HID, TPB), TPB, 0, stream>>>(agg1, denom, b1, NN, 6, 1);

  // ================= GAT layer 2 =================
  pack_B_f16<<<blks((F_HID / 32) * (F_OUT / 16) * 32, TPB), TPB, 0, stream>>>(
      W2, Bp2, F_HID / 32, F_OUT / 16, F_OUT);
  wmma_gemm_f16<F_HID, F_OUT / 16><<<NN / 32, 64, 0, stream>>>(agg1, Bp2, h2);
  node_scores<<<blks(NN, TPB), TPB, 0, stream>>>(h2, a2s, a2d, es, ed, NN, F_OUT);
  fill_u32<<<blks(NN, TPB), TPB, 0, stream>>>((unsigned*)mkey, 0x80000000u, NN);
  fill_u32<<<blks(NN, TPB), TPB, 0, stream>>>((unsigned*)denom, 0u, NN);
  fill_u32<<<blks((long long)NN * F_OUT, TPB), TPB, 0, stream>>>((unsigned*)agg2, 0u, NN * F_OUT);
  edge_pass1<<<blks(EP, TPB), TPB, 0, stream>>>(esrc, edst, es, ed, mkey, ep, NE, NN);
  edge_pass2<<<blks(EP, TPB), TPB, 0, stream>>>(edst, mkey, denom, ep, NE, NN);
  edge_agg<<<blks((long long)EP * F_OUT, TPB), TPB, 0, stream>>>(esrc, edst, ep, h2, agg2, NE, NN, 5);
  normalize_bias<<<blks((long long)NN * F_OUT, TPB), TPB, 0, stream>>>(agg2, denom, b2, NN, 5, 0);

  // ================= heads =================
  fill_u32<<<1, 32, 0, stream>>>((unsigned*)meanv, 0u, 32);
  col_sum32<<<256, 32, 0, stream>>>(agg2, meanv, NN);
  head_c<<<1, 32, 0, stream>>>(meanv, fc2_w1, fc2_b1, fc2_w2, fc2_b2, out, 1.0f / NN);
  lstm_seq<<<1, 128, 0, stream>>>(agg2, W_ih, W_hh, b_ih, b_hh, fc1_w, fc1_b, out, NN);
}
